// MambaResidualBlock_37452114821243
// MI455X (gfx1250) — matrix operand order repaired and checked by reading, not compiled, not run on hardware
//
#include <hip/hip_runtime.h>
#include <hip/hip_bf16.h>
#include <cstddef>
#include <cstdint>

// ---------------------------------------------------------------------------
// MI455X (gfx1250, wave32) Mamba residual block.
// GEMMs: v_wmma_f32_16x16x32_f16, double-buffered LDS A tile with
// global_load_async_to_lds_b128 (when available). Scan: 3-pass chunked
// parallel recurrence (32 chunks x 64 steps).
// ---------------------------------------------------------------------------

#define D_MODEL 1024
#define D_INNER 2048
#define D_STATE 16
#define D_CONV  4
#define DT_RANK 64
#define D_FF    4096
#define BATCH   2
#define SEQ     2048
#define BS      (BATCH * SEQ)   // 4096 tokens
#define NCHUNK  32
#define CLEN    (SEQ / NCHUNK)  // 64

typedef __attribute__((ext_vector_type(16))) _Float16 v16h;
typedef __attribute__((ext_vector_type(8)))  _Float16 v8h;
typedef __attribute__((ext_vector_type(8)))  float    v8f;
typedef int v4i __attribute__((vector_size(4 * sizeof(int))));   // matches builtin param

#define __AS1 __attribute__((address_space(1)))
#define __AS3 __attribute__((address_space(3)))

#if defined(__has_builtin)
#if __has_builtin(__builtin_amdgcn_global_load_async_to_lds_b128) && \
    __has_builtin(__builtin_amdgcn_s_wait_asynccnt)
#define HAVE_ASYNC_LDS 1
#endif
#endif

// ------------------------------ activations --------------------------------
__device__ __forceinline__ float act_silu(float x)     { return x / (1.0f + __expf(-x)); }
__device__ __forceinline__ float act_gelu(float x)     { return 0.5f * x * (1.0f + erff(x * 0.70710678118654752f)); }
__device__ __forceinline__ float act_softplus(float x) { return (x > 20.0f) ? x : log1pf(__expf(x)); }

// -------------------------- fp32 -> fp16 convert ---------------------------
__global__ void cvt_f32_to_f16(const float* __restrict__ src,
                               _Float16* __restrict__ dst, int n) {
  int i = blockIdx.x * blockDim.x + threadIdx.x;
  if (i < n) dst[i] = (_Float16)src[i];
}

// ------------------------------- layernorm ---------------------------------
__global__ __launch_bounds__(256) void layernorm_f16(
    const float* __restrict__ x, const float* __restrict__ w,
    const float* __restrict__ b, _Float16* __restrict__ out, int D) {
  __shared__ float s1[256];
  __shared__ float s2[256];
  const int row = blockIdx.x;
  const float* xr = x + (size_t)row * D;
  float sum = 0.f, sq = 0.f;
  for (int i = threadIdx.x; i < D; i += 256) {
    float v = xr[i];
    sum += v; sq += v * v;
  }
  s1[threadIdx.x] = sum; s2[threadIdx.x] = sq;
  __syncthreads();
  for (int s = 128; s > 0; s >>= 1) {
    if (threadIdx.x < s) { s1[threadIdx.x] += s1[threadIdx.x + s];
                           s2[threadIdx.x] += s2[threadIdx.x + s]; }
    __syncthreads();
  }
  const float mean = s1[0] / (float)D;
  const float var  = s2[0] / (float)D - mean * mean;
  const float rstd = rsqrtf(var + 1e-5f);
  _Float16* orow = out + (size_t)row * D;
  for (int i = threadIdx.x; i < D; i += 256)
    orow[i] = (_Float16)((xr[i] - mean) * rstd * w[i] + b[i]);
}

// ------------------------------ WMMA GEMM ----------------------------------
// C[M,N] = act(A[M,K] @ W[N,K]^T + bias) (+ residual).  A,W fp16 row-major.
// Block = 128x128 tile, 8 waves; wave owns 32x64 (2x4 WMMA accumulators).
// A tile (128 x BK=64) double-buffered in LDS, next tile fetched with
// async-to-LDS while current tile feeds 16 WMMAs.
// Requires: M % 128 == 0, K % 64 == 0, N % 16 == 0.
enum { EPI_NONE = 0, EPI_SOFTPLUS = 1, EPI_GELU = 2 };

template <int EPI, bool HAS_BIAS, bool HAS_RES>
__global__ __launch_bounds__(256, 2) void gemm_wmma_f16(
    const _Float16* __restrict__ A, const _Float16* __restrict__ W,
    const float* __restrict__ bias, const float* __restrict__ resid,
    float* __restrict__ Cf, _Float16* __restrict__ Ch,
    int M, int N, int K) {
  __shared__ __align__(32) _Float16 As[2][128 * 64];   // 2 x 16 KB

  const int tid  = threadIdx.x;
  const int wave = tid >> 5;
  const int lane = tid & 31;
  const int l15  = lane & 15;
  const int lhi  = lane >> 4;

  const int mblk = blockIdx.y * 128;
  const int nblk = blockIdx.x * 128;
  const int wm   = (wave >> 1) * 32;   // 0,32,64,96
  const int wn   = (wave & 1) * 64;    // 0,64

  // A-tile staging: thread moves one 64-byte run per stage.
  const int lrow = tid >> 1;           // 0..127
  const int lcol = (tid & 1) * 32;     // 0 or 32 halves

  auto stageA = [&](int buf, int k0) {
    const _Float16* src = A + (size_t)(mblk + lrow) * K + k0 + lcol;
    _Float16* dst = &As[buf][lrow * 64 + lcol];
#if defined(HAVE_ASYNC_LDS)
#pragma unroll
    for (int q = 0; q < 4; ++q)   // 4 x b128 = 64 bytes
      __builtin_amdgcn_global_load_async_to_lds_b128(
          (__AS1 v4i*)(uintptr_t)(src + q * 8),
          (__AS3 v4i*)(uint32_t)(uintptr_t)(dst + q * 8), 0, 0);
#else
    *(v16h*)dst        = *(const v16h*)src;
    *(v16h*)(dst + 16) = *(const v16h*)(src + 16);
#endif
  };

  v8f acc[2][4];
#pragma unroll
  for (int i = 0; i < 2; ++i)
#pragma unroll
    for (int j = 0; j < 4; ++j) acc[i][j] = (v8f){};

  const int nk = K / 64;
  stageA(0, 0);

  for (int it = 0; it < nk; ++it) {
    const int k0  = it * 64;
    const int cur = it & 1;
#if defined(HAVE_ASYNC_LDS)
    __builtin_amdgcn_s_wait_asynccnt(0);
#endif
    __syncthreads();
    if (it + 1 < nk) stageA(cur ^ 1, k0 + 64);   // overlap with WMMA below

#pragma unroll
    for (int kk = 0; kk < 64; kk += 32) {
      // ---- A fragments from LDS (ISA 16-bit A 16x32 layout) ----
      v16h afrag[2];
#pragma unroll
      for (int fm = 0; fm < 2; ++fm) {
        const int m = wm + fm * 16 + l15;
        v8h lo = *(const v8h*)&As[cur][m * 64 + kk + lhi * 8];
        v8h hi = *(const v8h*)&As[cur][m * 64 + kk + 16 + lhi * 8];
#pragma unroll
        for (int i = 0; i < 8; ++i) { afrag[fm][i] = lo[i]; afrag[fm][i + 8] = hi[i]; }
      }
      // ---- B fragments from global (weights are L2-resident) ----
      v16h bfrag[4];
#pragma unroll
      for (int fn = 0; fn < 4; ++fn) {
        int n = nblk + wn + fn * 16 + l15;
        int nc = (n < N) ? n : (N - 1);
        const _Float16* bgp = W + (size_t)nc * K + k0 + kk + lhi * 8;
        v8h blo = *(const v8h*)bgp;
        v8h bhi = *(const v8h*)(bgp + 16);
#pragma unroll
        for (int i = 0; i < 8; ++i) { bfrag[fn][i] = blo[i]; bfrag[fn][i + 8] = bhi[i]; }
        if (k0 + 64 < K) __builtin_prefetch(bgp + 64, 0, 1);
      }
      // ---- 8 WMMAs ----
#pragma unroll
      for (int fm = 0; fm < 2; ++fm)
#pragma unroll
        for (int fn = 0; fn < 4; ++fn)
          acc[fm][fn] = __builtin_amdgcn_wmma_f32_16x16x32_f16(
              false, afrag[fm], false, bfrag[fn],
              (short)0, acc[fm][fn], false, false);
    }
    __syncthreads();
  }

  // ---- epilogue: bias -> activation -> (+residual) -> store ----
#pragma unroll
  for (int fm = 0; fm < 2; ++fm) {
#pragma unroll
    for (int fn = 0; fn < 4; ++fn) {
      const int n = nblk + wn + fn * 16 + l15;
      if (n >= N) continue;                       // N % 16 == 0 -> wave-uniform
      const float bv = HAS_BIAS ? bias[n] : 0.0f;
#pragma unroll
      for (int r = 0; r < 8; ++r) {
        const int m = mblk + wm + fm * 16 + r + lhi * 8;
        float v = acc[fm][fn][r] + bv;
        if (EPI == EPI_SOFTPLUS) v = act_softplus(v);
        if (EPI == EPI_GELU)     v = act_gelu(v);
        if (HAS_RES)             v += resid[(size_t)m * N + n];
        if (Cf) Cf[(size_t)m * N + n] = v;
        if (Ch) Ch[(size_t)m * N + n] = (_Float16)v;
      }
    }
  }
}

// ----------------------- causal depthwise conv + SiLU ----------------------
__global__ void conv_silu(const float* __restrict__ xz,
                          const float* __restrict__ cw,
                          const float* __restrict__ cb,
                          float* __restrict__ u32,
                          _Float16* __restrict__ u16) {
  const long long i = (long long)blockIdx.x * blockDim.x + threadIdx.x;
  const long long total = (long long)BS * D_INNER;
  if (i >= total) return;
  const int d  = (int)(i & (D_INNER - 1));
  const int t  = (int)((i >> 11) & (SEQ - 1));
  const int bb = (int)(i >> 22);
  float acc = cb[d];
#pragma unroll
  for (int j = 0; j < D_CONV; ++j) {
    const int tt = t + j - (D_CONV - 1);
    if (tt >= 0)
      acc += cw[d * D_CONV + j] *
             xz[((size_t)bb * SEQ + tt) * (2 * D_INNER) + d];
  }
  const float v = act_silu(acc);
  u32[i] = v;
  u16[i] = (_Float16)v;
}

// ----------------- extract dt-rank slice of dbc as fp16 --------------------
__global__ void extract_dt(const float* __restrict__ dbc,
                           _Float16* __restrict__ dtin, int rows) {
  const int i = blockIdx.x * blockDim.x + threadIdx.x;
  if (i >= rows * DT_RANK) return;
  const int r = i >> 6, c = i & 63;
  dtin[i] = (_Float16)dbc[r * (DT_RANK + 2 * D_STATE) + c];
}

// --------------------- chunked selective scan: pass 1 ----------------------
// Per (b, chunk, d): run the 16-state recurrence from h=0 over CLEN steps,
// recording the chunk transfer: P = prod(alpha), R = h_end (zero start).
__global__ __launch_bounds__(256) void scan_pass1(
    const float* __restrict__ dlt, const float* __restrict__ u32,
    const float* __restrict__ dbc, const float* __restrict__ A_log,
    float* __restrict__ P, float* __restrict__ R) {
  const int i = blockIdx.x * blockDim.x + threadIdx.x;
  if (i >= BATCH * NCHUNK * D_INNER) return;
  const int d  = i & (D_INNER - 1);
  const int c  = (i >> 11) & (NCHUNK - 1);
  const int bb = i >> 16;

  float Av[D_STATE];
#pragma unroll
  for (int s = 0; s < D_STATE; ++s) Av[s] = -__expf(A_log[d * D_STATE + s]);

  float h[D_STATE], p[D_STATE];
#pragma unroll
  for (int s = 0; s < D_STATE; ++s) { h[s] = 0.0f; p[s] = 1.0f; }

  for (int t = 0; t < CLEN; ++t) {
    const size_t rt = (size_t)bb * SEQ + c * CLEN + t;
    const float dt = dlt[rt * D_INNER + d];
    const float du = dt * u32[rt * D_INNER + d];
    const float* bc = dbc + rt * (DT_RANK + 2 * D_STATE);
#pragma unroll
    for (int s = 0; s < D_STATE; ++s) {
      const float a = __expf(dt * Av[s]);
      p[s] *= a;
      h[s] = a * h[s] + du * bc[DT_RANK + s];
    }
  }
  const size_t base = (((size_t)bb * D_INNER + d) * NCHUNK + c) * D_STATE;
#pragma unroll
  for (int s = 0; s < D_STATE; ++s) { P[base + s] = p[s]; R[base + s] = h[s]; }
}

// --------------------- chunked selective scan: pass 2 ----------------------
// Per (b, d, s): compose chunk transfers sequentially (NCHUNK steps),
// emitting the true starting state of every chunk.
__global__ __launch_bounds__(256) void scan_pass2(
    const float* __restrict__ P, const float* __restrict__ R,
    float* __restrict__ Hs) {
  const int i = blockIdx.x * blockDim.x + threadIdx.x;
  if (i >= BATCH * D_INNER * D_STATE) return;
  const int s  = i & (D_STATE - 1);
  const int d  = (i >> 4) & (D_INNER - 1);
  const int bb = i >> 15;
  const size_t base = (((size_t)bb * D_INNER + d) * NCHUNK) * D_STATE + s;
  float h = 0.0f;
  for (int c = 0; c < NCHUNK; ++c) {
    const size_t o = base + (size_t)c * D_STATE;
    Hs[o] = h;
    h = P[o] * h + R[o];
  }
}

// --------------------- chunked selective scan: pass 3 ----------------------
// Per (b, chunk, d): replay the chunk from its true start state, produce
// y_t = h.C + u*D, fuse the SiLU(z) gate, emit fp16 for out_proj GEMM.
__global__ __launch_bounds__(256) void scan_pass3(
    const float* __restrict__ dlt, const float* __restrict__ u32,
    const float* __restrict__ xz,  const float* __restrict__ dbc,
    const float* __restrict__ A_log, const float* __restrict__ Dskip,
    const float* __restrict__ Hs, _Float16* __restrict__ g16) {
  const int i = blockIdx.x * blockDim.x + threadIdx.x;
  if (i >= BATCH * NCHUNK * D_INNER) return;
  const int d  = i & (D_INNER - 1);
  const int c  = (i >> 11) & (NCHUNK - 1);
  const int bb = i >> 16;

  float Av[D_STATE];
#pragma unroll
  for (int s = 0; s < D_STATE; ++s) Av[s] = -__expf(A_log[d * D_STATE + s]);
  const float Dv = Dskip[d];

  float h[D_STATE];
  const size_t base = (((size_t)bb * D_INNER + d) * NCHUNK + c) * D_STATE;
#pragma unroll
  for (int s = 0; s < D_STATE; ++s) h[s] = Hs[base + s];

  for (int t = 0; t < CLEN; ++t) {
    const size_t rt = (size_t)bb * SEQ + c * CLEN + t;
    const float dt = dlt[rt * D_INNER + d];
    const float ut = u32[rt * D_INNER + d];
    const float zt = xz[rt * (2 * D_INNER) + D_INNER + d];
    const float* bc = dbc + rt * (DT_RANK + 2 * D_STATE);
    const float du = dt * ut;
    float acc = 0.0f;
#pragma unroll
    for (int s = 0; s < D_STATE; ++s) {
      h[s] = __expf(dt * Av[s]) * h[s] + du * bc[DT_RANK + s];
      acc += h[s] * bc[DT_RANK + D_STATE + s];
    }
    const float y = acc + ut * Dv;
    g16[rt * D_INNER + d] = (_Float16)(y * act_silu(zt));
  }
}

// ---------------------------------------------------------------------------
extern "C" void kernel_launch(void* const* d_in, const int* in_sizes, int n_in,
                              void* d_out, int out_size, void* d_ws, size_t ws_size,
                              hipStream_t stream) {
  (void)in_sizes; (void)n_in; (void)out_size; (void)ws_size;

  const float* x        = (const float*)d_in[0];
  const float* ln1_w    = (const float*)d_in[1];
  const float* ln1_b    = (const float*)d_in[2];
  const float* in_proj  = (const float*)d_in[3];   // [4096,1024]
  const float* conv_w   = (const float*)d_in[4];   // [2048,1,4]
  const float* conv_b   = (const float*)d_in[5];
  const float* x_proj   = (const float*)d_in[6];   // [96,2048]
  const float* dt_w     = (const float*)d_in[7];   // [2048,64]
  const float* dt_b     = (const float*)d_in[8];
  const float* A_log    = (const float*)d_in[9];   // [2048,16]
  const float* Dskip    = (const float*)d_in[10];
  const float* out_proj = (const float*)d_in[11];  // [1024,2048]
  const float* ln2_w    = (const float*)d_in[12];
  const float* ln2_b    = (const float*)d_in[13];
  const float* ff1_w    = (const float*)d_in[14];  // [4096,1024]
  const float* ff1_b    = (const float*)d_in[15];
  const float* ff2_w    = (const float*)d_in[16];  // [1024,4096]
  const float* ff2_b    = (const float*)d_in[17];
  float* out = (float*)d_out;

  // -------- workspace bump allocator (256B-aligned blocks) --------
  char* ws = (char*)d_ws;
  size_t off = 0;
  auto alloc = [&](size_t bytes) -> char* {
    char* p = ws + off;
    off += (bytes + 255) & ~(size_t)255;
    return p;
  };
  _Float16* wh_in  = (_Float16*)alloc((size_t)4096 * 1024 * 2);
  _Float16* wh_xp  = (_Float16*)alloc((size_t)96 * 2048 * 2);
  _Float16* wh_dt  = (_Float16*)alloc((size_t)2048 * 64 * 2);
  _Float16* wh_op  = (_Float16*)alloc((size_t)1024 * 2048 * 2);
  _Float16* wh_f1  = (_Float16*)alloc((size_t)4096 * 1024 * 2);
  _Float16* wh_f2  = (_Float16*)alloc((size_t)1024 * 4096 * 2);
  _Float16* y1h    = (_Float16*)alloc((size_t)BS * D_MODEL * 2);
  float*    xz     = (float*)   alloc((size_t)BS * 2 * D_INNER * 4);
  float*    u32    = (float*)   alloc((size_t)BS * D_INNER * 4);
  _Float16* u16    = (_Float16*)alloc((size_t)BS * D_INNER * 2);
  float*    dbc    = (float*)   alloc((size_t)BS * 96 * 4);
  _Float16* dtin   = (_Float16*)alloc((size_t)BS * DT_RANK * 2);
  float*    dlt    = (float*)   alloc((size_t)BS * D_INNER * 4);
  _Float16* g16    = (_Float16*)alloc((size_t)BS * D_INNER * 2);
  float*    mo     = (float*)   alloc((size_t)BS * D_MODEL * 4);
  _Float16* y2h    = (_Float16*)alloc((size_t)BS * D_MODEL * 2);
  _Float16* h1h    = (_Float16*)alloc((size_t)BS * D_FF * 2);
  const size_t scan_sz = (size_t)BATCH * D_INNER * NCHUNK * D_STATE * 4; // 8MB
  float*    Pbuf   = (float*)alloc(scan_sz);
  float*    Rbuf   = (float*)alloc(scan_sz);
  float*    Hsbuf  = (float*)alloc(scan_sz);

  auto cvt = [&](const float* s, _Float16* d, int n) {
    cvt_f32_to_f16<<<(n + 255) / 256, 256, 0, stream>>>(s, d, n);
  };
  cvt(in_proj,  wh_in, 4096 * 1024);
  cvt(x_proj,   wh_xp, 96 * 2048);
  cvt(dt_w,     wh_dt, 2048 * 64);
  cvt(out_proj, wh_op, 1024 * 2048);
  cvt(ff1_w,    wh_f1, 4096 * 1024);
  cvt(ff2_w,    wh_f2, 1024 * 4096);

  // 1) y = LN1(x) -> fp16
  layernorm_f16<<<BS, 256, 0, stream>>>(x, ln1_w, ln1_b, y1h, D_MODEL);

  // 2) xz = y @ in_proj^T   [4096 x 4096, K=1024]
  gemm_wmma_f16<EPI_NONE, false, false><<<dim3(4096 / 128, BS / 128), 256, 0, stream>>>(
      y1h, wh_in, nullptr, nullptr, xz, nullptr, BS, 4096, 1024);

  // 3) u = silu(causal depthwise conv(xc) + conv_b)
  {
    long long total = (long long)BS * D_INNER;
    conv_silu<<<(unsigned)((total + 255) / 256), 256, 0, stream>>>(
        xz, conv_w, conv_b, u32, u16);
  }

  // 4) dbc = u @ x_proj^T   [4096 x 96, K=2048]
  gemm_wmma_f16<EPI_NONE, false, false><<<dim3(1, BS / 128), 256, 0, stream>>>(
      u16, wh_xp, nullptr, nullptr, dbc, nullptr, BS, 96, 2048);

  // 5) dlt = softplus(dbc[:, :64] @ dt_w^T + dt_b)  [4096 x 2048, K=64]
  extract_dt<<<(BS * DT_RANK + 255) / 256, 256, 0, stream>>>(dbc, dtin, BS);
  gemm_wmma_f16<EPI_SOFTPLUS, true, false><<<dim3(2048 / 128, BS / 128), 256, 0, stream>>>(
      dtin, wh_dt, dt_b, nullptr, dlt, nullptr, BS, 2048, 64);

  // 6) chunked selective scan (3 passes) + Dskip + silu(z) gate -> fp16
  {
    const int n13 = BATCH * NCHUNK * D_INNER;            // 131072
    const int n2  = BATCH * D_INNER * D_STATE;           // 65536
    scan_pass1<<<n13 / 256, 256, 0, stream>>>(dlt, u32, dbc, A_log, Pbuf, Rbuf);
    scan_pass2<<<n2 / 256, 256, 0, stream>>>(Pbuf, Rbuf, Hsbuf);
    scan_pass3<<<n13 / 256, 256, 0, stream>>>(dlt, u32, xz, dbc, A_log, Dskip,
                                              Hsbuf, g16);
  }

  // 7) mamba_out = g @ out_proj^T  [4096 x 1024, K=2048]
  gemm_wmma_f16<EPI_NONE, false, false><<<dim3(1024 / 128, BS / 128), 256, 0, stream>>>(
      g16, wh_op, nullptr, nullptr, mo, nullptr, BS, 1024, 2048);

  // 8) y2 = LN2(mamba_out) -> fp16
  layernorm_f16<<<BS, 256, 0, stream>>>(mo, ln2_w, ln2_b, y2h, D_MODEL);

  // 9) h = gelu(y2 @ ff1^T + ff1_b) -> fp16  [4096 x 4096, K=1024]
  gemm_wmma_f16<EPI_GELU, true, false><<<dim3(4096 / 128, BS / 128), 256, 0, stream>>>(
      y2h, wh_f1, ff1_b, nullptr, nullptr, h1h, BS, 4096, 1024);

  // 10) out = x + h @ ff2^T + ff2_b  [4096 x 1024, K=4096]
  gemm_wmma_f16<EPI_NONE, true, true><<<dim3(1024 / 128, BS / 128), 256, 0, stream>>>(
      h1h, wh_f2, ff2_b, x, out, nullptr, BS, 1024, 4096);
}
